// BetterAttendCompareAggregate_6133213299465
// MI455X (gfx1250) — matrix operand order repaired and checked by reading, not compiled, not run on hardware
//
#include <hip/hip_runtime.h>
#include <cstdint>
#include <cstddef>

#define B_   64
#define LQ   256
#define LP   256
#define DIM  1024

typedef __attribute__((ext_vector_type(16))) __bf16 v16bf;
typedef __attribute__((ext_vector_type(8)))  __bf16 v8bf;
typedef __attribute__((ext_vector_type(2)))  __bf16 v2bf;
typedef __attribute__((ext_vector_type(8)))  float  v8f;
typedef int v4i __attribute__((vector_size(16)));       // async builtin's pointee
typedef unsigned v4u __attribute__((vector_size(16)));
typedef __attribute__((ext_vector_type(4))) unsigned tdmv4u;
typedef __attribute__((ext_vector_type(8))) int      tdmv8i;
typedef __attribute__((ext_vector_type(4))) int      tdmv4i;

#if defined(__gfx1250__) && __has_builtin(__builtin_amdgcn_global_load_async_to_lds_b128)
#define HAVE_ASYNC 1
#else
#define HAVE_ASYNC 0
#endif
#if defined(__gfx1250__) && __has_builtin(__builtin_amdgcn_tensor_load_to_lds) && \
    __has_builtin(__builtin_amdgcn_s_wait_tensorcnt)
#define HAVE_TDM 1
#else
#define HAVE_TDM 0
#endif

#define AS1P(x) ((__attribute__((address_space(1))) v4i*)(void*)(x))
#define AS3P(x) ((__attribute__((address_space(3))) v4i*)(void*)(x))

__device__ __forceinline__ __bf16 f2bf(float f) {
  unsigned u = __builtin_bit_cast(unsigned, f);
  unsigned r = u + 0x7FFFu + ((u >> 16) & 1u);   // RNE
  unsigned short h = (unsigned short)(r >> 16);
  return __builtin_bit_cast(__bf16, h);
}

// pack two f32 into two bf16 (round-half-up) with one v_perm_b32
__device__ __forceinline__ unsigned pack_bf16_pair(float a, float b) {
  unsigned ua = __builtin_bit_cast(unsigned, a) + 0x8000u;
  unsigned ub = __builtin_bit_cast(unsigned, b) + 0x8000u;
  return __builtin_amdgcn_perm(ub, ua, 0x07060302u);
}

__device__ __forceinline__ void wait_async0() {
#if HAVE_ASYNC
#if __has_builtin(__builtin_amdgcn_s_wait_asynccnt)
  __builtin_amdgcn_s_wait_asynccnt(0);
#else
  asm volatile("s_wait_asynccnt 0" ::: "memory");
#endif
#endif
}

// 32-bit LDS byte offset of a __shared__ object (AS3 ptrtoint)
__device__ __forceinline__ unsigned lds_offset(const void* p) {
  return (unsigned)(size_t)((__attribute__((address_space(3))) const void*)p);
}

// Issue one TDM 2D tile DMA: H rows x W elems, row stride in elems, into LDS.
// D# layout per CDNA5 ISA (2D tile: groups 2/3 zero,
// tile_dim2 = 0). dsz_code: 0=1B,1=2B,2=4B,3=8B.
__device__ __forceinline__ void tdm_load_2d(unsigned lds_off, const void* gaddr,
                                            unsigned dsz_code, unsigned W, unsigned H,
                                            unsigned long long stride_elems) {
#if HAVE_TDM
  unsigned long long ga = (unsigned long long)gaddr;
  tdmv4u g0;
  g0[0] = 1u;                                      // count=1, user descriptor
  g0[1] = lds_off;                                 // lds_addr
  g0[2] = (unsigned)(ga & 0xFFFFFFFFu);            // global_addr[31:0]
  g0[3] = (unsigned)((ga >> 32) & 0x1FFFFFFu)      // global_addr[56:32]
        | (2u << 30);                              // type = 2 ("image")
  tdmv8i g1;
  g1[0] = (int)(dsz_code << 16);                   // wg_mask=0 | data_size
  g1[1] = (int)((W & 0xFFFFu) << 16);              // tensor_dim0[15:0] @bit48
  g1[2] = (int)((W >> 16) | ((H & 0xFFFFu) << 16));// tensor_dim0[31:16] | tensor_dim1[15:0]
  g1[3] = (int)((H >> 16) | ((W & 0xFFFFu) << 16));// tensor_dim1[31:16] | tile_dim0
  g1[4] = (int)(H & 0xFFFFu);                      // tile_dim1 | tile_dim2=0 (2D)
  g1[5] = (int)(unsigned)(stride_elems & 0xFFFFFFFFull);      // dim0_stride[31:0]
  g1[6] = (int)(unsigned)((stride_elems >> 32) & 0xFFFFull);  // dim0_stride[47:32]
  g1[7] = 0;
  tdmv4i gz  = {0, 0, 0, 0};
  tdmv8i gz8 = {0, 0, 0, 0, 0, 0, 0, 0};
  __builtin_amdgcn_tensor_load_to_lds(g0, g1, gz, gz, gz8, 0);  // 6-arg form
#else
  (void)lds_off; (void)gaddr; (void)dsz_code; (void)W; (void)H; (void)stride_elems;
#endif
}

// ---------------- elementwise convert: f32 -> bf16 -------------------------
__global__ __launch_bounds__(256) void f32_to_bf16_k(const float* __restrict__ x,
                                                     __bf16* __restrict__ y, long n) {
  long i = (long)blockIdx.x * 256 + threadIdx.x;
  if (i < n) y[i] = f2bf(x[i]);
}

// ---------------- WMMA NT GEMM: C[M,N] = A[M,K] * B[N,K]^T -----------------
template <typename AT> struct raw8;
template <> struct raw8<float>  { using t = v8f; };
template <> struct raw8<__bf16> { using t = v8bf; };

__device__ __forceinline__ v8bf to_bf(v8bf v) { return v; }
__device__ __forceinline__ v8bf to_bf(v8f v) {
  v4u w = { pack_bf16_pair(v[0], v[1]), pack_bf16_pair(v[2], v[3]),
            pack_bf16_pair(v[4], v[5]), pack_bf16_pair(v[6], v[7]) };
  return __builtin_bit_cast(v8bf, w);
}
__device__ __forceinline__ void stc(float* p, float v)  { *p = v; }
__device__ __forceinline__ void stc(__bf16* p, float v) { *p = f2bf(v); }

// Block = 256 threads = 8 waves; block tile 64(M) x 128(N); wave tile 16x64.
// K-step 32. A/B tiles double-buffered in LDS. Preferred fill path: one TDM
// tensor_load_to_lds per tile, issued by wave 0 (TENSORcnt + barrier handoff).
// Fallbacks: per-lane async-to-LDS (ASYNCcnt), then plain copies.
template <typename AT, typename OT>
__global__ __launch_bounds__(256) void wmma_gemm_nt(
    const AT* __restrict__ A, const __bf16* __restrict__ Bm, OT* __restrict__ C,
    int K, long sA, int lda, long sB, int ldb, long sC, int ldc) {
  __shared__ AT     Asm[2][64 * 32];
  __shared__ __bf16 Bsm[2][128 * 32];

  const int lane = threadIdx.x & 31;
  const int wave = threadIdx.x >> 5;
  const int b    = blockIdx.z;
  const int bm0  = blockIdx.y * 64;
  const int bn0  = blockIdx.x * 128;
  const int hi   = lane >> 4;           // 0: lanes 0-15, 1: lanes 16-31
  const int l16  = lane & 15;

  const AT*     Abase = A  + (long)b * sA + (long)bm0 * lda;
  const __bf16* Bbase = Bm + (long)b * sB + (long)bn0 * ldb;
  OT*           Cb    = C  + (long)b * sC;

  // stage one K-step (32 wide) of A (64 rows) and B (128 rows) into LDS[buf]
  auto stage = [&](int buf, int k0) {
#if HAVE_TDM
    if (wave == 0) {
      tdm_load_2d(lds_offset(&Asm[buf][0]), Abase + k0,
                  sizeof(AT) == 4 ? 2u : 1u, 32u, 64u, (unsigned long long)lda);
      tdm_load_2d(lds_offset(&Bsm[buf][0]), Bbase + k0,
                  1u, 32u, 128u, (unsigned long long)ldb);
    }
#else
    constexpr int CH  = 16 / sizeof(AT);   // elems per 16B chunk
    constexpr int CPR = 32 / CH;           // chunks per A row
    for (int ch = threadIdx.x; ch < 64 * CPR; ch += 256) {
      const int r = ch / CPR, c = ch % CPR;
      const AT* src = Abase + (long)r * lda + k0 + c * CH;
      AT* dst = &Asm[buf][r * 32 + c * CH];
#if HAVE_ASYNC
      __builtin_amdgcn_global_load_async_to_lds_b128(AS1P(src), AS3P(dst), 0, 0);
#else
      *(v4i*)dst = *(const v4i*)src;
#endif
    }
    for (int ch = threadIdx.x; ch < 128 * 4; ch += 256) {   // B rows are 4x16B
      const int r = ch >> 2, c = ch & 3;
      const __bf16* src = Bbase + (long)r * ldb + k0 + c * 8;
      __bf16* dst = &Bsm[buf][r * 32 + c * 8];
#if HAVE_ASYNC
      __builtin_amdgcn_global_load_async_to_lds_b128(AS1P(src), AS3P(dst), 0, 0);
#else
      *(v4i*)dst = *(const v4i*)src;
#endif
    }
#endif
  };
  auto stage_wait = [&]() {
#if HAVE_TDM
    if (wave == 0) __builtin_amdgcn_s_wait_tensorcnt(0);
#else
    wait_async0();
#endif
  };

  using R = typename raw8<AT>::t;
  v8f acc0 = {}, acc1 = {}, acc2 = {}, acc3 = {};

  stage(0, 0);
  stage_wait();
  __syncthreads();

  int buf = 0;
#pragma unroll 1
  for (int k0 = 0; k0 < K; k0 += 32) {
    if (k0 + 32 < K) stage(buf ^ 1, k0 + 32);   // DMA fill of next buffer

    // A 16x32 bf16 frag: lane<16 K=[0..15]; lane>=16 K=[16..31] (matches B)
    const AT* asrc = &Asm[buf][((wave & 3) * 16 + l16) * 32];
    v8bf alo = to_bf(*(const R*)(asrc + hi * 16));
    v8bf ahi = to_bf(*(const R*)(asrc + hi * 16 + 8));
    v16bf afrag = __builtin_shufflevector(alo, ahi,
                                          0,1,2,3,4,5,6,7,8,9,10,11,12,13,14,15);
    // B 32x16 frags: lane%16 = N, 16 contiguous K at (lane>=16)*16
    const __bf16* bsrc = &Bsm[buf][(((wave >> 2) * 64) + l16) * 32 + hi * 16];
    v16bf b0 = *(const v16bf*)(bsrc);
    v16bf b1 = *(const v16bf*)(bsrc + 16 * 32);
    v16bf b2 = *(const v16bf*)(bsrc + 32 * 32);
    v16bf b3 = *(const v16bf*)(bsrc + 48 * 32);
    acc0 = __builtin_amdgcn_wmma_f32_16x16x32_bf16(false, afrag, false, b0, (short)0, acc0, false, false);
    acc1 = __builtin_amdgcn_wmma_f32_16x16x32_bf16(false, afrag, false, b1, (short)0, acc1, false, false);
    acc2 = __builtin_amdgcn_wmma_f32_16x16x32_bf16(false, afrag, false, b2, (short)0, acc2, false, false);
    acc3 = __builtin_amdgcn_wmma_f32_16x16x32_bf16(false, afrag, false, b3, (short)0, acc3, false, false);

    stage_wait();         // issuing wave's DMA of buf^1 complete
    __syncthreads();      // everyone done reading buf / LDS of buf^1 published
    buf ^= 1;
  }

  // C/D 16x16 f32 layout: N = lane%16, M = (lane>=16 ? 8 : 0) + vgpr index
  const int m0 = bm0 + (wave & 3) * 16;
  const int n0 = bn0 + (wave >> 2) * 64;
  const int mb = m0 + hi * 8;
#pragma unroll
  for (int r = 0; r < 8; ++r) {
    OT* crow = Cb + (long)(mb + r) * ldc + n0 + l16;
    stc(crow,      acc0[r]);
    stc(crow + 16, acc1[r]);
    stc(crow + 32, acc2[r]);
    stc(crow + 48, acc3[r]);
  }
}

// ---------------- masked softmax + column-sums -----------------------------
__device__ __forceinline__ float block_red_max(float v, float* red8) {
  for (int m = 16; m > 0; m >>= 1) v = fmaxf(v, __shfl_xor(v, m, 32));
  if ((threadIdx.x & 31) == 0) red8[threadIdx.x >> 5] = v;
  __syncthreads();
  float r = red8[0];
#pragma unroll
  for (int i = 1; i < 8; ++i) r = fmaxf(r, red8[i]);
  __syncthreads();
  return r;
}
__device__ __forceinline__ float block_red_sum(float v, float* red8) {
  for (int m = 16; m > 0; m >>= 1) v += __shfl_xor(v, m, 32);
  if ((threadIdx.x & 31) == 0) red8[threadIdx.x >> 5] = v;
  __syncthreads();
  float r = red8[0];
#pragma unroll
  for (int i = 1; i < 8; ++i) r += red8[i];
  __syncthreads();
  return r;
}

// One block per batch. Phase 1: cj[b,p] = sum_q softmax_j[b,q,p] (rows staged
// 8 at a time into LDS via async copy). Phase 2: ci[b,q] = sum_p softmax_i[b,p,q].
__global__ __launch_bounds__(256) void masked_softmax_colsum(
    const float* __restrict__ e, const int* __restrict__ qm, const int* __restrict__ pm,
    float* __restrict__ cj, float* __restrict__ ci) {
  __shared__ float red8[8];
  __shared__ float rows[8 * 256];
  const int b = blockIdx.x, t = threadIdx.x;
  const float* eb = e + (long)b * (LQ * LP);
  const float pmv = (float)pm[b * LP + t];
  const float qmv = (float)qm[b * LQ + t];

  // ---- phase 1: thread owns column p = t ----
  float cjacc = 0.f;
  for (int q0 = 0; q0 < LQ; q0 += 8) {
#if HAVE_ASYNC
    {
      const float* src = eb + q0 * LP + t * 8;   // 8 rows = 8KB, 32B per thread
      float* dst = &rows[t * 8];
      __builtin_amdgcn_global_load_async_to_lds_b128(AS1P(src), AS3P(dst), 0, 0);
      __builtin_amdgcn_global_load_async_to_lds_b128(AS1P(src + 4), AS3P(dst + 4), 0, 0);
      wait_async0();
    }
    __syncthreads();
#else
    for (int i = t; i < 8 * 256; i += 256) rows[i] = eb[q0 * LP + i];
    __syncthreads();
#endif
#pragma unroll 1
    for (int r = 0; r < 8; ++r) {
      const int q = q0 + r;
      const float m  = (float)qm[b * LQ + q] * pmv;
      const float x  = rows[r * 256 + t] * m;
      const float mx = block_red_max(x, red8);
      const float ex = __expf(x - mx) * m;
      const float sm = block_red_sum(ex, red8);
      cjacc += ex / (sm + 1e-6f);
    }
    __syncthreads();   // rows reuse barrier
  }
  cj[b * LP + t] = cjacc;

  // ---- phase 2: thread owns column q = t of e^T ----
  float ciacc = 0.f;
#pragma unroll 1
  for (int p = 0; p < LP; ++p) {
    const float m  = qmv * (float)pm[b * LP + p];
    const float x  = eb[(long)t * LP + p] * m;
    const float mx = block_red_max(x, red8);
    const float ex = __expf(x - mx) * m;
    const float sm = block_red_sum(ex, red8);
    ciacc += ex / (sm + 1e-6f);
  }
  ci[b * LQ + t] = ciacc;
}

// ---------------- weighted sums over L ------------------------------------
// s1[b] = [ sum_l q[l,b,:]  ;  sum_p cj[b,p] * p[p,b,:] ]
// s2[b] = [ sum_l p[l,b,:]  ;  sum_q ci[b,q] * q[q,b,:] ]
__global__ __launch_bounds__(256) void weighted_sums(
    const float* __restrict__ q, const float* __restrict__ p,
    const float* __restrict__ cj, const float* __restrict__ ci,
    float* __restrict__ s1, float* __restrict__ s2) {
  const int d = blockIdx.x * 256 + threadIdx.x;
  const int b = blockIdx.y;
  float sq = 0.f, sp = 0.f, sb = 0.f, sa = 0.f;
  for (int l = 0; l < LQ; ++l) {
    const float qv = q[((long)l * B_ + b) * DIM + d];
    const float pv = p[((long)l * B_ + b) * DIM + d];
    sq += qv;                     sp += pv;
    sb += cj[b * LP + l] * pv;    sa += ci[b * LQ + l] * qv;
  }
  s1[(long)b * (2 * DIM) + d]       = sq;
  s1[(long)b * (2 * DIM) + DIM + d] = sb;
  s2[(long)b * (2 * DIM) + d]       = sp;
  s2[(long)b * (2 * DIM) + DIM + d] = sa;
}

// ---------------- collapse WG,WH into two 2048-vectors ---------------------
__global__ __launch_bounds__(256) void make_wvec(const float* __restrict__ WG,
                                                 const float* __restrict__ WH,
                                                 float* __restrict__ wv) {
  const int k = blockIdx.x * 256 + threadIdx.x;   // 0..2047
  float a = 0.f, c = 0.f;
  for (int n = 0; n < DIM; ++n) {
    const float g = WG[(long)n * (2 * DIM) + k];
    a += WH[n] * g;
    c += WH[DIM + n] * g;
  }
  wv[k] = a;
  wv[2 * DIM + k] = c;
}

// ---------------- final dot products --------------------------------------
__global__ __launch_bounds__(256) void final_out(const float* __restrict__ s1,
                                                 const float* __restrict__ s2,
                                                 const float* __restrict__ wv,
                                                 float* __restrict__ out) {
  __shared__ float red8[8];
  const int b = blockIdx.x, t = threadIdx.x;
  float acc = 0.f;
  for (int k = t; k < 2 * DIM; k += 256)
    acc += s1[(long)b * (2 * DIM) + k] * wv[k] +
           s2[(long)b * (2 * DIM) + k] * wv[2 * DIM + k];
  float r = block_red_sum(acc, red8);
  if (t == 0) out[b] = r;
}

// ===========================================================================
extern "C" void kernel_launch(void* const* d_in, const int* in_sizes, int n_in,
                              void* d_out, int out_size, void* d_ws, size_t ws_size,
                              hipStream_t stream) {
  (void)in_sizes; (void)n_in; (void)out_size; (void)ws_size;
  const float* q  = (const float*)d_in[0];   // [Lq,B,D]
  const float* p  = (const float*)d_in[1];   // [Lp,B,D]
  const int*   qm = (const int*)d_in[2];     // [B,Lq]
  const int*   pm = (const int*)d_in[3];     // [B,Lp]
  const float* WF = (const float*)d_in[4];   // [D,D]
  const float* WG = (const float*)d_in[5];   // [D,2D]
  const float* WH = (const float*)d_in[6];   // [1,2D]
  float* out = (float*)d_out;                // [B,1]

  char* wsp = (char*)d_ws;
  auto alloc = [&](size_t bytes) -> char* {
    char* r = wsp; wsp += (bytes + 255) & ~(size_t)255; return r;
  };
  __bf16* WFb  = (__bf16*)alloc(sizeof(__bf16) * DIM * DIM);             // 2 MB
  __bf16* attq = (__bf16*)alloc(sizeof(__bf16) * (size_t)B_ * LQ * DIM); // 33.5 MB
  __bf16* attp = (__bf16*)alloc(sizeof(__bf16) * (size_t)B_ * LP * DIM); // 33.5 MB
  float*  e    = (float*)alloc(sizeof(float) * (size_t)B_ * LQ * LP);    // 16.8 MB
  float*  cj   = (float*)alloc(sizeof(float) * B_ * LP);
  float*  ci   = (float*)alloc(sizeof(float) * B_ * LQ);
  float*  s1   = (float*)alloc(sizeof(float) * B_ * 2 * DIM);
  float*  s2   = (float*)alloc(sizeof(float) * B_ * 2 * DIM);
  float*  wv   = (float*)alloc(sizeof(float) * 2 * 2 * DIM);

  // 1) WF -> bf16
  f32_to_bf16_k<<<(DIM * DIM) / 256, 256, 0, stream>>>(WF, WFb, (long)DIM * DIM);

  // 2) projections: att = X @ WF^T per batch. A = q[:,b,:] (f32, lda=B*D),
  //    B = WF_bf16 (N=K=D), C = att[b] (bf16, L x D).
  dim3 g1(DIM / 128, LQ / 64, B_);
  wmma_gemm_nt<float, __bf16><<<g1, 256, 0, stream>>>(
      q, WFb, attq, DIM, (long)DIM, B_ * DIM, 0L, DIM, (long)LQ * DIM, DIM);
  wmma_gemm_nt<float, __bf16><<<g1, 256, 0, stream>>>(
      p, WFb, attp, DIM, (long)DIM, B_ * DIM, 0L, DIM, (long)LP * DIM, DIM);

  // 3) e[b] = attq[b] @ attp[b]^T  (f32 out, 256x256, K=1024)
  dim3 g2(LP / 128, LQ / 64, B_);
  wmma_gemm_nt<__bf16, float><<<g2, 256, 0, stream>>>(
      attq, attp, e, DIM, (long)LQ * DIM, DIM, (long)LP * DIM, DIM, (long)LQ * LP, LP);

  // 4) masked softmax column-sums in both directions
  masked_softmax_colsum<<<B_, 256, 0, stream>>>(e, qm, pm, cj, ci);

  // 5) L-sums and attention-weighted sums (exploits sum/linear commutation)
  weighted_sums<<<dim3(DIM / 256, B_), 256, 0, stream>>>(q, p, cj, ci, s1, s2);

  // 6) fold WG into WH (out = s1 . (WH[:D] @ WG) + s2 . (WH[D:] @ WG))
  make_wvec<<<(2 * DIM) / 256, 256, 0, stream>>>(WG, WH, wv);

  // 7) final per-batch dot products
  final_out<<<B_, 256, 0, stream>>>(s1, s2, wv, out);
}
